// Embedder_39797166965440
// MI455X (gfx1250) — matrix operand order repaired and checked by reading, not compiled, not run on hardware
//
#include <hip/hip_runtime.h>

#define S_TOT 4096
#define E_DIM 1024
#define H_NUM 16
#define D_DIM 64

typedef __attribute__((ext_vector_type(16))) _Float16 v16h;
typedef __attribute__((ext_vector_type(16))) __bf16   v16bf;
typedef __attribute__((ext_vector_type(8)))  __bf16   v8bf;
typedef __attribute__((ext_vector_type(8)))  float    v8f;

static __device__ __forceinline__ __bf16 f32_to_bf16(float f) {
  union { float f; unsigned u; } a; a.f = f;
  unsigned u = a.u + 0x7FFFu + ((a.u >> 16) & 1u);
  unsigned short h = (unsigned short)(u >> 16);
  __bf16 r;
  __builtin_memcpy(&r, &h, sizeof(r));
  return r;
}

// ---------------- K1a: partial column sums of x (HBM-dominant: prefetch ahead) ----------------
__global__ void k_colsum_partial(const float* __restrict__ x, float* __restrict__ csPart) {
  int e = blockIdx.x * 256 + threadIdx.x;
  int c = blockIdx.y;
  int s0 = c * (S_TOT / 16);
  float acc = 0.f;
  for (int s = s0; s < s0 + S_TOT / 16; ++s) {
    if (s + 8 < s0 + S_TOT / 16)
      __builtin_prefetch(&x[(long)(s + 8) * E_DIM + e], 0, 1);
    acc += x[(long)s * E_DIM + e];
  }
  csPart[(long)c * E_DIM + e] = acc;
}

// ---------------- K1b: finalize column sums ----------------
__global__ void k_colsum_final(const float* __restrict__ csPart, float* __restrict__ cs) {
  int e = blockIdx.x * 256 + threadIdx.x;
  float acc = 0.f;
  for (int c = 0; c < 16; ++c) acc += csPart[(long)c * E_DIM + e];
  cs[e] = acc;
}

// ---------------- K2: find segment range of pos ----------------
__global__ void k_segrange(const int* __restrict__ seg, const int* __restrict__ posp,
                           int* __restrict__ segInfo) {
  __shared__ int smin, smax;
  if (threadIdx.x == 0) { smin = S_TOT; smax = -1; }
  __syncthreads();
  int target = seg[posp[0]];
  for (int i = threadIdx.x; i < S_TOT; i += 256) {
    if (seg[i] == target) { atomicMin(&smin, i); atomicMax(&smax, i); }
  }
  __syncthreads();
  if (threadIdx.x == 0) { segInfo[0] = smin; segInfo[1] = smax + 1; }
}

// ---------------- K3: vsum[j] = colsum(x) . Wv[j,:] + S*bv[j] ----------------
__global__ void k_vsum(const float* __restrict__ cs, const float* __restrict__ Wv,
                       const float* __restrict__ bv, float* __restrict__ vsum) {
  int lane = threadIdx.x & 31, wid = threadIdx.x >> 5;
  int j = blockIdx.x * 8 + wid;
  const float* wr = Wv + (long)j * E_DIM;
  float acc = 0.f;
  for (int e = lane; e < E_DIM; e += 32) acc += cs[e] * wr[e];
  for (int m = 16; m >= 1; m >>= 1) acc += __shfl_xor(acc, m, 32);
  if (lane == 0) vsum[j] = acc + (float)S_TOT * bv[j];
}

// ---------------- K4: WMMA f16 QKV projection for active row tiles ----------------
// q16/k16: row-major [S,E] f16;  vT: transposed [E,S] bf16 (for contiguous-t PV B-frags)
__global__ void k_qkv_proj(const float* __restrict__ x,
                           const float* __restrict__ Wq, const float* __restrict__ bq,
                           const float* __restrict__ Wk, const float* __restrict__ bk,
                           const float* __restrict__ Wv, const float* __restrict__ bv,
                           const int* __restrict__ segInfo,
                           _Float16* __restrict__ q16, _Float16* __restrict__ k16,
                           __bf16* __restrict__ vT) {
  int rt = blockIdx.x;
  int s0 = segInfo[0], s1 = segInfo[1];
  int r0 = rt * 16;
  // attention reads a 32-aligned-down expansion of the segment; cover it
  int lo = (s0 & ~31);
  if (!(r0 < s1 && r0 + 16 > lo)) return;

  int z = blockIdx.z;
  const float* W    = (z == 0) ? Wq : (z == 1) ? Wk : Wv;
  const float* bias = (z == 0) ? bq : (z == 1) ? bk : bv;

  int lane = threadIdx.x & 31, wid = threadIdx.x >> 5;
  int n0 = blockIdx.y * 128 + wid * 16;
  int m = r0 + (lane & 15);
  int n = n0 + (lane & 15);
  int hi = (lane & 16) ? 1 : 0;
  int ka = hi ? 16 : 0;   // A-fragment K base (contiguous k = 16h + s)
  int kb = hi ? 16 : 0;   // B-fragment K base (contiguous k = 16h + s)
  const float* xr = x + (long)m * E_DIM;
  const float* wr = W + (long)n * E_DIM;

  v8f c = {};
  for (int k0 = 0; k0 < E_DIM; k0 += 32) {
    v16h a, b;
    const float* pa = xr + k0 + ka;
    const float* pb = wr + k0 + kb;
#pragma unroll
    for (int j = 0; j < 8; ++j) {
      a[j]     = (_Float16)pa[j];
      a[8 + j] = (_Float16)pa[8 + j];
    }
#pragma unroll
    for (int j = 0; j < 16; ++j) b[j] = (_Float16)pb[j];
    c = __builtin_amdgcn_wmma_f32_16x16x32_f16(false, a, false, b, (short)0, c, false, false);
  }

  float bn = bias[n];
  if (z == 2) {
    // 8 consecutive rows of column n in vT -> one 16B store
    v8bf pack;
#pragma unroll
    for (int r = 0; r < 8; ++r) pack[r] = f32_to_bf16(c[r] + bn);
    *(v8bf*)(vT + (long)n * S_TOT + r0 + hi * 8) = pack;
  } else {
    _Float16* drow = ((z == 0) ? q16 : k16) + (long)(r0 + hi * 8) * E_DIM + n;
#pragma unroll
    for (int r = 0; r < 8; ++r) drow[(long)r * E_DIM] = (_Float16)(c[r] + bn);
  }
}

// ---------------- K5: in-segment attention (f16 QK^T WMMA + bf16 PV WMMA) ----------------
// Writes per-(rowtile,head) column-sums of wv over in-segment rows into partial[rt][E].
__global__ void k_attn(const _Float16* __restrict__ q16, const _Float16* __restrict__ k16,
                       const __bf16* __restrict__ vT, const float* __restrict__ vsum,
                       const int* __restrict__ segInfo, float* __restrict__ partial) {
  int rt = blockIdx.x, h = blockIdx.y;
  int lane = threadIdx.x;  // 32-thread (one-wave) block
  int s0 = segInfo[0], s1 = segInfo[1];
  int r0 = rt * 16;
  float* myPart = partial + (long)rt * E_DIM + h * D_DIM;
  if (!(r0 < s1 && r0 + 16 > s0)) {
    myPart[lane] = 0.f;
    myPart[lane + 32] = 0.f;
    return;
  }

  __shared__ __bf16 P[16][32];
  __bf16 bz = f32_to_bf16(0.0f);
  int hi = (lane & 16) ? 1 : 0;
  int m = r0 + (lane & 15);

  // Q fragments (A layout, f16): K = D = 64 -> two chunks of 32
  v16h aq0, aq1;
  {
    const _Float16* qr = q16 + (long)m * E_DIM + h * D_DIM;
    int ka = hi ? 16 : 0;
#pragma unroll
    for (int j = 0; j < 8; ++j) {
      aq0[j]     = qr[ka + j];
      aq0[8 + j] = qr[ka + 8 + j];
      aq1[j]     = qr[32 + ka + j];
      aq1[8 + j] = qr[32 + ka + 8 + j];
    }
  }

  v8f acc0 = {}, acc1 = {}, acc2 = {}, acc3 = {};
  float den[8];
#pragma unroll
  for (int r = 0; r < 8; ++r) den[r] = 0.f;

  for (int t0 = (s0 & ~31); t0 < s1; t0 += 32) {
    // two 16-wide score tiles -> e = exp(score)-1 (0 outside segment) staged in LDS
#pragma unroll
    for (int half = 0; half < 2; ++half) {
      int t = t0 + half * 16 + (lane & 15);
      int kb = hi ? 16 : 0;
      const _Float16* kr = k16 + (long)t * E_DIM + h * D_DIM + kb;
      v16h b0, b1;
#pragma unroll
      for (int j = 0; j < 16; ++j) { b0[j] = kr[j]; b1[j] = kr[32 + j]; }
      v8f sc = {};
      sc = __builtin_amdgcn_wmma_f32_16x16x32_f16(false, aq0, false, b0, (short)0, sc, false, false);
      sc = __builtin_amdgcn_wmma_f32_16x16x32_f16(false, aq1, false, b1, (short)0, sc, false, false);
      bool tin = (t >= s0) && (t < s1);
      int col = half * 16 + (lane & 15);
#pragma unroll
      for (int r = 0; r < 8; ++r) {
        float e = tin ? (__expf(sc[r]) - 1.0f) : 0.0f;
        den[r] += e;
        P[r + hi * 8][col] = f32_to_bf16(e);
      }
    }
    // P fragment (A layout, bf16)
    v16bf pf;
    {
      int mr = lane & 15;
      int ka = hi ? 16 : 0;
#pragma unroll
      for (int j = 0; j < 8; ++j) {
        pf[j]     = P[mr][ka + j];
        pf[8 + j] = P[mr][ka + 8 + j];
      }
    }
    int kb3 = hi ? 16 : 0;
#pragma unroll
    for (int dt = 0; dt < 4; ++dt) {
      int ecol = h * D_DIM + dt * 16 + (lane & 15);
      const __bf16* vr = vT + (long)ecol * S_TOT + t0 + kb3;
      v16bf vb;
#pragma unroll
      for (int j = 0; j < 16; ++j) {
        int t = t0 + kb3 + j;
        vb[j] = (t >= s0 && t < s1) ? vr[j] : bz;   // guard: never mix garbage into PV
      }
      v8f* accp = (dt == 0) ? &acc0 : (dt == 1) ? &acc1 : (dt == 2) ? &acc2 : &acc3;
      *accp = __builtin_amdgcn_wmma_f32_16x16x32_bf16(false, pf, false, vb, (short)0, *accp, false, false);
    }
  }

  // reduce den across the 16 lanes of each half-group; Z = sum(e) + S
#pragma unroll
  for (int r = 0; r < 8; ++r) {
    den[r] += __shfl_xor(den[r], 1, 32);
    den[r] += __shfl_xor(den[r], 2, 32);
    den[r] += __shfl_xor(den[r], 4, 32);
    den[r] += __shfl_xor(den[r], 8, 32);
    den[r] += (float)S_TOT;
  }

  // wv = (acc + vsum) / Z ; column-sum over in-segment rows of this tile
#pragma unroll
  for (int dt = 0; dt < 4; ++dt) {
    v8f acc = (dt == 0) ? acc0 : (dt == 1) ? acc1 : (dt == 2) ? acc2 : acc3;
    int ecol = h * D_DIM + dt * 16 + (lane & 15);
    float vs = vsum[ecol];
    float colp = 0.f;
#pragma unroll
    for (int r = 0; r < 8; ++r) {
      int mm = r0 + r + hi * 8;
      float wv = (acc[r] + vs) / den[r];
      colp += (mm >= s0 && mm < s1) ? wv : 0.f;
    }
    colp += __shfl_xor(colp, 16, 32);  // lanes L and L+16 hold the same column
    if (lane < 16) myPart[dt * 16 + lane] = colp;
  }
}

// ---------------- K6: wvsum[e] = sum over row tiles (deterministic order) ----------------
__global__ void k_wvsum(const float* __restrict__ partial, float* __restrict__ wvsum) {
  int e = blockIdx.x * 256 + threadIdx.x;
  float acc = 0.f;
  for (int rt = 0; rt < S_TOT / 16; ++rt) acc += partial[(long)rt * E_DIM + e];
  wvsum[e] = acc;
}

// ---------------- K7: result[n] = wvsum . Wo[n,:] / C + bo[n] ----------------
__global__ void k_result(const float* __restrict__ wvsum, const float* __restrict__ Wo,
                         const float* __restrict__ bo, const int* __restrict__ segInfo,
                         float* __restrict__ out) {
  int lane = threadIdx.x & 31, wid = threadIdx.x >> 5;
  int n = blockIdx.x * 8 + wid;
  const float* wr = Wo + (long)n * E_DIM;
  float acc = 0.f;
  for (int e = lane; e < E_DIM; e += 32) acc += wvsum[e] * wr[e];
  for (int m = 16; m >= 1; m >>= 1) acc += __shfl_xor(acc, m, 32);
  if (lane == 0) {
    float invC = 1.0f / (float)(segInfo[1] - segInfo[0]);
    out[n] = acc * invC + bo[n];
  }
}

extern "C" void kernel_launch(void* const* d_in, const int* in_sizes, int n_in,
                              void* d_out, int out_size, void* d_ws, size_t ws_size,
                              hipStream_t stream) {
  const float* x  = (const float*)d_in[0];
  const float* Wq = (const float*)d_in[1];
  const float* bq = (const float*)d_in[2];
  const float* Wk = (const float*)d_in[3];
  const float* bk = (const float*)d_in[4];
  const float* Wv = (const float*)d_in[5];
  const float* bv = (const float*)d_in[6];
  const float* Wo = (const float*)d_in[7];
  const float* bo = (const float*)d_in[8];
  const int*  seg = (const int*)d_in[9];
  const int* posp = (const int*)d_in[10];
  float* out = (float*)d_out;

  char* w = (char*)d_ws;
  int*   segInfo = (int*)w;                                  // 8 B (pad 256)
  float* csPart  = (float*)(w + 256);                        // 16*1024 f32
  float* cs      = csPart + 16 * E_DIM;                      // 1024 f32
  float* vsum    = cs + E_DIM;                               // 1024 f32
  float* partial = vsum + E_DIM;                             // 256*1024 f32
  float* wvsum   = partial + (S_TOT / 16) * E_DIM;           // 1024 f32
  _Float16* q16  = (_Float16*)(wvsum + E_DIM);               // S*E f16
  _Float16* k16  = q16 + (long)S_TOT * E_DIM;                // S*E f16
  __bf16*   vT   = (__bf16*)(k16 + (long)S_TOT * E_DIM);     // E*S bf16

  k_colsum_partial<<<dim3(E_DIM / 256, 16), 256, 0, stream>>>(x, csPart);
  k_colsum_final<<<dim3(E_DIM / 256), 256, 0, stream>>>(csPart, cs);
  k_segrange<<<dim3(1), 256, 0, stream>>>(seg, posp, segInfo);
  k_vsum<<<dim3(E_DIM / 8), 256, 0, stream>>>(cs, Wv, bv, vsum);
  k_qkv_proj<<<dim3(S_TOT / 16, E_DIM / 128, 3), 256, 0, stream>>>(
      x, Wq, bq, Wk, bk, Wv, bv, segInfo, q16, k16, vT);
  k_attn<<<dim3(S_TOT / 16, H_NUM), 32, 0, stream>>>(q16, k16, vT, vsum, segInfo, partial);
  k_wvsum<<<dim3(E_DIM / 256), 256, 0, stream>>>(partial, wvsum);
  k_result<<<dim3(E_DIM / 8), 256, 0, stream>>>(wvsum, Wo, bo, segInfo, out);
}
